// AdaptiveTokenFilter_51445118271913
// MI455X (gfx1250) — matrix operand order repaired and checked by reading, not compiled, not run on hardware
//
#include <hip/hip_runtime.h>

#define B_DIM 8
#define S_DIM 4096
#define E_DIM 1024
#define H_DIM 128
#define MIN_K 32
#define BS_STRIDE 40  // bf16 elems per LDS row of B tile (32 data + 8 pad = 80 bytes)

typedef __attribute__((ext_vector_type(16))) __bf16 v16bf;
typedef __attribute__((ext_vector_type(8)))  float  v8f;

// ---------------------------------------------------------------------------
// K0: w1 [E,H] fp32 row-major  ->  w1t [H,E] bf16 row-major (K contiguous)
// ---------------------------------------------------------------------------
__global__ __launch_bounds__(256) void k_convert_w1(const float* __restrict__ w1,
                                                    __bf16* __restrict__ w1t) {
  int i = blockIdx.x * 256 + threadIdx.x;   // 0 .. H*E-1
  int k = i & (E_DIM - 1);                  // E index
  int n = i >> 10;                          // H index
  w1t[i] = (__bf16)w1[k * H_DIM + n];
}

// ---------------------------------------------------------------------------
// K1: fused scorer: logits[t] = relu(X[t,:]@W1 + b1) @ w2 + b2
// wave = 16 tokens; K-loop over 1024 in steps of 32; N=128 -> 8 bf16 WMMAs/step.
// Double-buffered LDS B tile: one barrier per K-step, next tile staged in the
// shadow of the current WMMA chain. B fragments in private regs for ILP.
// ---------------------------------------------------------------------------
__global__ __launch_bounds__(256) void k_scorer(
    const float* __restrict__ x, const __bf16* __restrict__ w1t,
    const float* __restrict__ b1, const float* __restrict__ w2,
    const float* __restrict__ b2, float* __restrict__ logits)
{
  __shared__ __bf16 Bs[2][H_DIM * BS_STRIDE];
  const int tid  = threadIdx.x;
  const int wave = tid >> 5;
  const int lane = tid & 31;
  const int hf   = lane >> 4;   // which 16-lane half
  const int l16  = lane & 15;
  const long m0  = (long)blockIdx.x * 128 + wave * 16;

  // staging indices (thread t stages row n = t/2, K half kh = t&1)
  const int sn = tid >> 1;
  const int skh = tid & 1;
  const __bf16* srcbase = w1t + sn * E_DIM + skh * 16;
  __bf16* dst0 = &Bs[0][sn * BS_STRIDE + skh * 16];
  __bf16* dst1 = &Bs[1][sn * BS_STRIDE + skh * 16];

  v8f acc[8] = {};

  // A-fragment rows: lane -> token m0+l16; one 16-elem K run at hf*16
  const float* arow = x + (m0 + l16) * E_DIM + hf * 16;

  // stage first B tile (k0 = 0) into buffer 0
  {
    const uint4* s4 = (const uint4*)srcbase;
    uint4 v0 = s4[0], v1 = s4[1];
    uint4* d4 = (uint4*)dst0;
    d4[0] = v0; d4[1] = v1;
  }

  for (int it = 0; it < E_DIM / 32; ++it) {
    const int k0  = it * 32;
    const int buf = it & 1;

    // issue A loads (fp32) for this step before the barrier
    const float4* p = (const float4*)(arow + k0);
    float4 f0 = p[0], f1 = p[1];
    const float4* q = (const float4*)(arow + k0 + 8);
    float4 f2 = q[0], f3 = q[1];

    __syncthreads();   // current buffer fully staged; previous buffer fully read

    // stage next B tile into the other buffer (overlaps with WMMA below)
    if (it + 1 < E_DIM / 32) {
      const uint4* s4 = (const uint4*)(srcbase + (k0 + 32));
      uint4 v0 = s4[0], v1 = s4[1];
      uint4* d4 = (uint4*)(buf ? dst0 : dst1);
      d4[0] = v0; d4[1] = v1;
    }

    // convert A to bf16 fragment
    v16bf a;
    a[0]=(__bf16)f0.x;  a[1]=(__bf16)f0.y;  a[2]=(__bf16)f0.z;  a[3]=(__bf16)f0.w;
    a[4]=(__bf16)f1.x;  a[5]=(__bf16)f1.y;  a[6]=(__bf16)f1.z;  a[7]=(__bf16)f1.w;
    a[8]=(__bf16)f2.x;  a[9]=(__bf16)f2.y;  a[10]=(__bf16)f2.z; a[11]=(__bf16)f2.w;
    a[12]=(__bf16)f3.x; a[13]=(__bf16)f3.y; a[14]=(__bf16)f3.z; a[15]=(__bf16)f3.w;

    // load all 8 B fragments into private regs (independent ds_load_b128 pairs)
    v16bf bfrag[8];
#pragma unroll
    for (int f = 0; f < 8; ++f) {
      int n = f * 16 + l16;   // B col for this lane
      union { uint4 u[2]; v16bf v; } bb;
      const uint4* bp = (const uint4*)(&Bs[buf][n * BS_STRIDE + hf * 16]);
      bb.u[0] = bp[0];
      bb.u[1] = bp[1];
      bfrag[f] = bb.v;
    }

#pragma unroll
    for (int f = 0; f < 8; ++f) {
#ifdef __HIP_DEVICE_COMPILE__
      acc[f] = __builtin_amdgcn_wmma_f32_16x16x32_bf16(
          /*neg_a=*/false, a, /*neg_b=*/false, bfrag[f],
          /*c_mod=*/(short)0, acc[f], /*reuse_a=*/false, /*reuse_b=*/false);
#endif
    }
  }

  // epilogue: relu + dot with w2 (lane holds cols n = 16f+l16, rows r+8*hf)
  float s[8] = {0.f,0.f,0.f,0.f,0.f,0.f,0.f,0.f};
#pragma unroll
  for (int f = 0; f < 8; ++f) {
    int n = f * 16 + l16;
    float bb = b1[n];
    float ww = w2[n];
#pragma unroll
    for (int r = 0; r < 8; ++r) {
      float h = acc[f][r] + bb;
      h = h > 0.f ? h : 0.f;
      s[r] += h * ww;
    }
  }
#pragma unroll
  for (int r = 0; r < 8; ++r) {
    s[r] += __shfl_xor(s[r], 1, 32);
    s[r] += __shfl_xor(s[r], 2, 32);
    s[r] += __shfl_xor(s[r], 4, 32);
    s[r] += __shfl_xor(s[r], 8, 32);
  }
  if (l16 == 0) {
    float bias2 = b2[0];
#pragma unroll
    for (int r = 0; r < 8; ++r)
      logits[m0 + hf * 8 + r] = s[r] + bias2;
  }
}

// ---------------------------------------------------------------------------
// K2: per batch row: expected_k, k = clamp(int(expected_k), 32),
//     z = logit - log(-log(u)), hard top-k mask via 4-pass radix select.
// ---------------------------------------------------------------------------
__global__ __launch_bounds__(1024) void k_select(
    const float* __restrict__ logits, const float* __restrict__ u,
    float* __restrict__ mask, float* __restrict__ expk)
{
  __shared__ unsigned keys[S_DIM];
  __shared__ float red[32];
  __shared__ unsigned hist[256];
  __shared__ unsigned sh_rem, sh_prefix, sh_eq;

  const int b   = blockIdx.x;
  const int tid = threadIdx.x;
  const float* lrow = logits + (size_t)b * S_DIM;
  const float* urow = u + (size_t)b * S_DIM;

  float ssum = 0.f;
  for (int s = tid; s < S_DIM; s += 1024) {
    float lg = lrow[s];
    ssum += 1.f / (1.f + expf(-lg));
    float z = lg - logf(-logf(urow[s]));   // TAU == 1
    unsigned bits = __float_as_uint(z);
    // monotone float->uint transform (descending select on uint)
    keys[s] = (bits & 0x80000000u) ? ~bits : (bits | 0x80000000u);
  }
  ssum += __shfl_xor(ssum, 16, 32);
  ssum += __shfl_xor(ssum, 8, 32);
  ssum += __shfl_xor(ssum, 4, 32);
  ssum += __shfl_xor(ssum, 2, 32);
  ssum += __shfl_xor(ssum, 1, 32);
  if ((tid & 31) == 0) red[tid >> 5] = ssum;
  __syncthreads();
  if (tid == 0) {
    float v = 0.f;
    for (int w = 0; w < 32; ++w) v += red[w];
    expk[b] = v;
    int k = (int)v;
    if (k < MIN_K) k = MIN_K;
    if (k > S_DIM) k = S_DIM;
    sh_rem = (unsigned)k;
    sh_eq = 0;
    sh_prefix = 0;
  }

  unsigned prefMask = 0, prefix = 0;
  for (int pass = 0; pass < 4; ++pass) {
    const int shift = 24 - pass * 8;
    __syncthreads();
    if (tid < 256) hist[tid] = 0;
    __syncthreads();
    for (int s = tid; s < S_DIM; s += 1024) {
      unsigned key = keys[s];
      if ((key & prefMask) == prefix)
        atomicAdd(&hist[(key >> shift) & 255u], 1u);
    }
    __syncthreads();
    if (tid == 0) {
      unsigned rem = sh_rem;
      unsigned cum = 0;
      int dsel = 0;
      for (int d = 255; d >= 0; --d) {
        unsigned c = hist[d];
        if (cum + c >= rem) { dsel = d; rem -= cum; break; }
        cum += c;
      }
      sh_rem = rem;  // how many keys equal to the threshold are still needed
      sh_prefix = prefix | ((unsigned)dsel << shift);
    }
    __syncthreads();
    prefix = sh_prefix;
    prefMask |= (0xFFu << shift);
  }
  const unsigned thrKey = prefix;   // exact k-th largest key
  const unsigned needEq = sh_rem;   // #keys == thrKey to keep
  __syncthreads();

  float* mrow = mask + (size_t)b * S_DIM;
  for (int s = tid; s < S_DIM; s += 1024) {
    unsigned key = keys[s];
    float m = 0.f;
    if (key > thrKey) m = 1.f;
    else if (key == thrKey) {
      if (atomicAdd(&sh_eq, 1u) < needEq) m = 1.f;
    }
    mrow[s] = m;
  }
}

// ---------------------------------------------------------------------------
// K3: filtered = x * mask (float4 per thread; 256 float4 per token)
// ---------------------------------------------------------------------------
__global__ __launch_bounds__(256) void k_apply(
    const float* __restrict__ x, const float* __restrict__ mask,
    float* __restrict__ out)
{
  long i = (long)blockIdx.x * 256 + threadIdx.x;   // float4 index
  const float4 v = ((const float4*)x)[i];
  const float m = mask[i >> 8];                    // E/4 = 256 float4 per token
  float4 o;
  o.x = v.x * m; o.y = v.y * m; o.z = v.z * m; o.w = v.w * m;
  ((float4*)out)[i] = o;
}

// ---------------------------------------------------------------------------
extern "C" void kernel_launch(void* const* d_in, const int* in_sizes, int n_in,
                              void* d_out, int out_size, void* d_ws, size_t ws_size,
                              hipStream_t stream) {
  const float* x  = (const float*)d_in[0];  // [B,S,E]
  const float* w1 = (const float*)d_in[1];  // [E,H]
  const float* b1 = (const float*)d_in[2];  // [H]
  const float* w2 = (const float*)d_in[3];  // [H,1]
  const float* b2 = (const float*)d_in[4];  // [1]
  const float* u  = (const float*)d_in[5];  // [B,S]

  __bf16* w1t   = (__bf16*)d_ws;                                     // 256 KB
  float* logits = (float*)((char*)d_ws + (size_t)H_DIM * E_DIM * 2); // 128 KB

  float* out_filtered = (float*)d_out;                                // [B,S,E]
  float* out_mask     = out_filtered + (size_t)B_DIM * S_DIM * E_DIM; // [B,S]
  float* out_expk     = out_mask + (size_t)B_DIM * S_DIM;             // [B]

  k_convert_w1<<<(H_DIM * E_DIM) / 256, 256, 0, stream>>>(w1, w1t);
  k_scorer<<<(B_DIM * S_DIM) / 128, 256, 0, stream>>>(x, w1t, b1, w2, b2, logits);
  k_select<<<B_DIM, 1024, 0, stream>>>(logits, u, out_mask, out_expk);
  k_apply<<<(int)((size_t)B_DIM * S_DIM * E_DIM / 4 / 256), 256, 0, stream>>>(
      x, out_mask, out_filtered);
}
